// GCALoss_83013127897103
// MI455X (gfx1250) — matrix operand order repaired and checked by reading, not compiled, not run on hardware
//
#include <hip/hip_runtime.h>

typedef __attribute__((ext_vector_type(16))) _Float16     v16h;
typedef __attribute__((ext_vector_type(8)))  float        v8f;
typedef __attribute__((ext_vector_type(4))) unsigned int  u32x4;
typedef __attribute__((ext_vector_type(8)))  int          i32x8;
typedef __attribute__((ext_vector_type(4)))  int          i32x4;

#define NROWS    8192
#define DDIM     128
#define M2       (2 * NROWS)     // 16384 stacked rows
#define TEMP_INV 2.5f            // 1 / 0.4
// exp(x/temp) == exp2(x * (1/temp)*log2(e)); args bounded by |x|<=1 -> safe
#define EXP2_SCALE 3.6067376022224087f

#define WG_ROWS  128             // rows per workgroup  (A panel)
#define WG_COLS  256             // cols per workgroup  (B panel)

#if __has_builtin(__builtin_amdgcn_tensor_load_to_lds)
#define HAVE_TDM 1
#else
#define HAVE_TDM 0
#endif

#if __has_builtin(__builtin_amdgcn_exp2f)
#define FAST_EXP2(x) __builtin_amdgcn_exp2f(x)
#else
#define FAST_EXP2(x) __expf((x) * 0.6931471805599453f)
#endif

// ---------------------------------------------------------------------------
// Zero the row-sum accumulator and the output scalar.
// ---------------------------------------------------------------------------
__global__ void gca_init_kernel(float* __restrict__ rowsum, float* __restrict__ outp) {
    int i = blockIdx.x * blockDim.x + threadIdx.x;
    if (i < M2) rowsum[i] = 0.0f;
    if (i == 0) *outp = 0.0f;
}

// ---------------------------------------------------------------------------
// L2-normalize each row of [z1; z2] and store as f16 row-major [16384, 128].
// ---------------------------------------------------------------------------
__global__ void gca_normalize_kernel(const float* __restrict__ z1,
                                     const float* __restrict__ z2,
                                     _Float16* __restrict__ Zh) {
    __shared__ float red[DDIM];
    const int r = blockIdx.x;
    const int t = threadIdx.x;
    const float* src = (r < NROWS) ? (z1 + (size_t)r * DDIM)
                                   : (z2 + (size_t)(r - NROWS) * DDIM);
    float x = src[t];
    red[t] = x * x;
    __syncthreads();
    for (int s = 64; s > 0; s >>= 1) {
        if (t < s) red[t] += red[t + s];
        __syncthreads();
    }
    float inv = 1.0f / fmaxf(sqrtf(red[0]), 1e-12f);
    Zh[(size_t)r * DDIM + t] = (_Float16)(x * inv);
}

#if HAVE_TDM
// ---------------------------------------------------------------------------
// Issue one TDM descriptor: load `nrows` rows x 256B from global into LDS.
// 2D tile, data_size = 8B (32 elements/row), contiguous (stride = 32).
// ---------------------------------------------------------------------------
__device__ __forceinline__ void tdm_load_panel(unsigned int lds_off,
                                               const _Float16* gptr,
                                               int nrows) {
    unsigned long long ga = (unsigned long long)(uintptr_t)gptr;
    u32x4 g0;
    g0[0] = 1u;                                   // count=1 (valid), user mode
    g0[1] = lds_off;                              // lds_addr (bytes)
    g0[2] = (unsigned int)(ga & 0xFFFFFFFFu);     // global_addr[31:0]
    g0[3] = (unsigned int)((ga >> 32) & 0x1FFFFFFu) | (2u << 30); // addr hi | type=2
    i32x8 g1;
    g1[0] = (3 << 16);                            // mask=0, data_size=3 (8B)
    g1[1] = (32 << 16);                           // tensor_dim0 = 32 (bits 63:48)
    g1[2] = ((nrows & 0xFFFF) << 16);             // tensor_dim1 lo (bits 95:80)
    g1[3] = ((nrows >> 16) & 0xFFFF) | (32 << 16);// tensor_dim1 hi | tile_dim0 = 32
    g1[4] = (nrows & 0xFFFF);                     // tile_dim1 = nrows, tile_dim2 = 0
    g1[5] = 32;                                   // tensor_dim0_stride = 32 (8B units)
    g1[6] = 0;
    g1[7] = 0;
    i32x4 z4 = {};
#if defined(__clang_major__) && (__clang_major__ >= 23)
    i32x8 z8 = {};
    __builtin_amdgcn_tensor_load_to_lds(g0, g1, z4, z4, z8, 0);
#else
    __builtin_amdgcn_tensor_load_to_lds(g0, g1, z4, z4, 0);
#endif
}
#endif

// ---------------------------------------------------------------------------
// Fused Gram + exp + row-sum.
// Block = 256 threads = 8 wave32 (4 row-waves x 2 col-waves).
// WG tile: 128 rows x 256 cols, both panels TDM-staged in LDS (96 KB).
// Wave tile: 32 rows x (4 iters x 32 cols); 64 WMMAs per wave.
// ---------------------------------------------------------------------------
__global__ __launch_bounds__(256)
void gca_gram_rowsum_kernel(const _Float16* __restrict__ Zh,
                            float* __restrict__ rowsum) {
    extern __shared__ _Float16 smem[];
    _Float16* sA = smem;                     // WG_ROWS * DDIM  (32 KB)
    _Float16* sB = smem + WG_ROWS * DDIM;    // WG_COLS * DDIM  (64 KB)

    const int lane = threadIdx.x & 31;
    const int w    = threadIdx.x >> 5;
    const int wrow = w & 3;                  // 0..3
    const int wcol = w >> 2;                 // 0..1
    const int m    = lane & 15;
    const int hi   = lane >> 4;

    const int R  = blockIdx.x * WG_ROWS;     // global row base of A panel
    const int CB = blockIdx.y * WG_COLS;     // global row base of B panel

    // ---- Stage both panels into LDS via the Tensor Data Mover.
#if HAVE_TDM
    if (threadIdx.x < 32) {                  // one wave issues both DMAs
        tdm_load_panel((unsigned int)(uintptr_t)sA, Zh + (size_t)R  * DDIM, WG_ROWS);
        tdm_load_panel((unsigned int)(uintptr_t)sB, Zh + (size_t)CB * DDIM, WG_COLS);
        __builtin_amdgcn_s_wait_tensorcnt(0);
    }
#else
    {
        const u32x4* gA = (const u32x4*)(Zh + (size_t)R  * DDIM);
        const u32x4* gB = (const u32x4*)(Zh + (size_t)CB * DDIM);
        u32x4* lA = (u32x4*)sA;
        u32x4* lB = (u32x4*)sB;
        for (int i = threadIdx.x; i < WG_ROWS * 16; i += 256) lA[i] = gA[i];
        for (int i = threadIdx.x; i < WG_COLS * 16; i += 256) lB[i] = gB[i];
    }
#endif
    __syncthreads();

    // ---- A fragments (hoisted for all K): af[mt][ks], rows wrow*32 + mt*16.
    // ISA 16-bit A 16x32 layout: lane m (half hi) holds K = hi*8 + [0..7]
    // and K = 16 + hi*8 + [0..7]  -> two 16-byte LDS reads per fragment.
    v16h af[2][4];
#pragma unroll
    for (int mt = 0; mt < 2; ++mt) {
        const _Float16* row = sA + (size_t)(wrow * 32 + mt * 16 + m) * DDIM;
#pragma unroll
        for (int ks = 0; ks < 4; ++ks) {
            const int kb = ks * 32;
            u32x4* dst = (u32x4*)&af[mt][ks];
            dst[0] = *(const u32x4*)(row + kb + hi * 8);
            dst[1] = *(const u32x4*)(row + kb + 16 + hi * 8);
        }
    }

    v8f sumacc[2] = {};    // per-row partial sum of exp, elementwise

    for (int ci = 0; ci < 4; ++ci) {
        const int Cl = (ci * 2 + wcol) * 32;       // col offset inside B panel

        // ---- All B fragments for this 32-col block up front (one clause).
        // ISA 32x16 B layout: column n = lane&15, K = hi*8 + [0..7]
        // and K = 16 + hi*8 + [0..7] -> two 16-byte LDS reads per fragment.
        v16h b[2][4];                              // [nt][ks]
#pragma unroll
        for (int nt = 0; nt < 2; ++nt) {
            const _Float16* col = sB + (size_t)(Cl + nt * 16 + m) * DDIM + hi * 8;
#pragma unroll
            for (int ks = 0; ks < 4; ++ks) {
                u32x4* dst = (u32x4*)&b[nt][ks];
                dst[0] = *(const u32x4*)(col + ks * 32);
                dst[1] = *(const u32x4*)(col + ks * 32 + 16);
            }
        }

        v8f d[2][2] = {};
#pragma unroll
        for (int ks = 0; ks < 4; ++ks)
#pragma unroll
            for (int mt = 0; mt < 2; ++mt)
#pragma unroll
                for (int nt = 0; nt < 2; ++nt)
                    d[mt][nt] = __builtin_amdgcn_wmma_f32_16x16x32_f16(
                        false, af[mt][ks], false, b[nt][ks],
                        (short)0, d[mt][nt], false, false);

        // fused exp(G/temp) = exp2(G * EXP2_SCALE): one multiply + v_exp_f32
#pragma unroll
        for (int mt = 0; mt < 2; ++mt)
#pragma unroll
            for (int e = 0; e < 8; ++e)
                sumacc[mt][e] += FAST_EXP2(d[mt][0][e] * EXP2_SCALE)
                               + FAST_EXP2(d[mt][1][e] * EXP2_SCALE);
    }

    // ---- Reduce 16 columns per half-wave, then one atomic per row.
    // C/D layout: VGPR v -> row v (lanes 0-15) / row v+8 (lanes 16-31).
#pragma unroll
    for (int mt = 0; mt < 2; ++mt)
#pragma unroll
        for (int v = 0; v < 8; ++v) {
            float x = sumacc[mt][v];
            x += __shfl_xor(x, 1, 32);
            x += __shfl_xor(x, 2, 32);
            x += __shfl_xor(x, 4, 32);
            x += __shfl_xor(x, 8, 32);
            if ((lane & 15) == 0) {
                const int row = R + wrow * 32 + mt * 16 + v + hi * 8;
                atomicAdd(&rowsum[row], x);
            }
        }
}

// ---------------------------------------------------------------------------
// Finalize: pos_i = dot(z1n_i, z2n_i);
// loss_i = -pos_i/temp + 0.5*(log(rs[i]-e^2.5) + log(rs[i+N]-e^2.5))
// ---------------------------------------------------------------------------
__global__ void gca_finalize_kernel(const _Float16* __restrict__ Zh,
                                    const float* __restrict__ rowsum,
                                    float* __restrict__ outp) {
    __shared__ float red[DDIM];
    const int i = blockIdx.x;
    const int t = threadIdx.x;
    float a = (float)Zh[(size_t)i * DDIM + t];
    float b = (float)Zh[(size_t)(i + NROWS) * DDIM + t];
    red[t] = a * b;
    __syncthreads();
    for (int s = 64; s > 0; s >>= 1) {
        if (t < s) red[t] += red[t + s];
        __syncthreads();
    }
    if (t == 0) {
        const float E  = __expf(TEMP_INV);     // exp(1/temp): refl diagonal
        const float d1 = rowsum[i] - E;
        const float d2 = rowsum[i + NROWS] - E;
        const float term = -red[0] * TEMP_INV + 0.5f * (__logf(d1) + __logf(d2));
        atomicAdd(outp, term * (1.0f / (float)NROWS));
    }
}

// ---------------------------------------------------------------------------
extern "C" void kernel_launch(void* const* d_in, const int* in_sizes, int n_in,
                              void* d_out, int out_size, void* d_ws, size_t ws_size,
                              hipStream_t stream) {
    const float* z1 = (const float*)d_in[0];   // out      [8192,128] f32
    const float* z2 = (const float*)d_in[1];   // aug_out  [8192,128] f32
    float* outp = (float*)d_out;               // scalar f32

    _Float16* Zh     = (_Float16*)d_ws;                              // 4 MB
    float*    rowsum = (float*)((char*)d_ws +
                        (size_t)M2 * DDIM * sizeof(_Float16));       // 64 KB

    gca_init_kernel<<<(M2 + 255) / 256, 256, 0, stream>>>(rowsum, outp);
    gca_normalize_kernel<<<M2, DDIM, 0, stream>>>(z1, z2, Zh);

    dim3 grid(M2 / WG_ROWS, M2 / WG_COLS);     // 128 x 64 = 8192 workgroups
    size_t lds_bytes = (size_t)(WG_ROWS + WG_COLS) * DDIM * sizeof(_Float16); // 96 KB
    gca_gram_rowsum_kernel<<<grid, 256, lds_bytes, stream>>>(Zh, rowsum);

    gca_finalize_kernel<<<NROWS, DDIM, 0, stream>>>(Zh, rowsum, outp);
}
